// DeformableDecoderLayer_1151051235429
// MI455X (gfx1250) — matrix operand order repaired and checked by reading, not compiled, not run on hardware
//
#include <hip/hip_runtime.h>
#include <hip/hip_bf16.h>

#define BQ 8
#define NQV 1000
#define DM 256
#define NHH 8
#define HDD 32
#define LLV 4
#define PPN 4
#define DFFN 1024
#define STOT 21760
#define NQP 1024
#define M1 (BQ * NQV)      // 8000 query rows
#define MV (BQ * STOT)     // 174080 value rows

typedef __attribute__((ext_vector_type(16))) __bf16 bf16x16;
typedef __attribute__((ext_vector_type(8)))  float  f32x8;
typedef __attribute__((ext_vector_type(4)))  unsigned int u32x4;
typedef __attribute__((ext_vector_type(4)))  int       i32x4;

union FragAB { bf16x16 v; u32x4 q[2]; unsigned short u[16]; };

__device__ __forceinline__ unsigned short f2bf(float f) {
  unsigned int u = __float_as_uint(f);
  unsigned int r = u + 0x7fffu + ((u >> 16) & 1u);
  return (unsigned short)(r >> 16);
}
__device__ __forceinline__ float bf2f(unsigned short h) {
  return __uint_as_float(((unsigned int)h) << 16);
}

#if __has_builtin(__builtin_amdgcn_global_load_async_to_lds_b128) && \
    __has_builtin(__builtin_amdgcn_s_wait_asynccnt)
#define HAS_ASYNC_LDS 1
typedef __attribute__((address_space(1))) i32x4 GAS1;   // global v4i32
typedef __attribute__((address_space(3))) i32x4 LAS3;   // LDS v4i32
#else
#define HAS_ASYNC_LDS 0
#endif

// ---------------------------------------------------------------------------
// Generic bf16 WMMA GEMM: C[M,N] = A[M,K] @ Wt[N,K]^T + bias, optional ReLU.
// Block 256 thr = 8 waves (4x2), tile 128x128, K-step 32.
// A/B tiles double-buffered in LDS via async global->LDS copies (ASYNCcnt).
// ---------------------------------------------------------------------------
template<int RELU, int WRITE_F32, int WRITE_BF16>
__global__ __launch_bounds__(256)
void gemm_bf16_wmma(const unsigned short* __restrict__ A,
                    const unsigned short* __restrict__ Wt,
                    const float* __restrict__ bias,
                    float* __restrict__ Cf, unsigned short* __restrict__ Cb,
                    int M, int N, int K)
{
  __shared__ unsigned short Atile[2][128 * 32];   // 8KB each
  __shared__ unsigned short Btile[2][128 * 32];

  const int lane = threadIdx.x & 31;
  const int wave = threadIdx.x >> 5;
  const int wr = wave >> 1;             // 0..3
  const int wc = wave & 1;              // 0..1
  const int rb0 = blockIdx.y * 128;
  const int cb0 = blockIdx.x * 128;
  const int lm = lane & 15;
  const int kh = lane >> 4;

  f32x8 acc[2][4];
#pragma unroll
  for (int mi = 0; mi < 2; ++mi)
#pragma unroll
    for (int ni = 0; ni < 4; ++ni)
      acc[mi][ni] = (f32x8){0.f,0.f,0.f,0.f,0.f,0.f,0.f,0.f};

  // stage one 128x32 A tile + one 128x32 B tile (16B chunks, 4 issues/thread)
  auto stage = [&](int kb, int which) {
#pragma unroll
    for (int t = 0; t < 2; ++t) {
      int c = threadIdx.x + t * 256;    // 0..511
      int row = c >> 2, sub = c & 3;    // row 0..127, 16B chunk 0..3
      int ar = rb0 + row; if (ar >= M) ar = M - 1;
      const unsigned short* ag = A  + (size_t)ar * K + kb + sub * 8;
      const unsigned short* bg = Wt + (size_t)(cb0 + row) * K + kb + sub * 8;
      unsigned short* al = &Atile[which][row * 32 + sub * 8];
      unsigned short* bl = &Btile[which][row * 32 + sub * 8];
#if HAS_ASYNC_LDS
      __builtin_amdgcn_global_load_async_to_lds_b128(
          (GAS1*)(i32x4*)ag, (LAS3*)(i32x4*)al, 0, 0);
      __builtin_amdgcn_global_load_async_to_lds_b128(
          (GAS1*)(i32x4*)bg, (LAS3*)(i32x4*)bl, 0, 0);
#else
      *(u32x4*)al = *(const u32x4*)ag;
      *(u32x4*)bl = *(const u32x4*)bg;
#endif
    }
  };

  int buf = 0;
  stage(0, 0);
  for (int kb = 0; kb < K; kb += 32) {
    const bool more = (kb + 32) < K;
    if (more) stage(kb + 32, buf ^ 1);
#if HAS_ASYNC_LDS
    if (more) __builtin_amdgcn_s_wait_asynccnt(4);   // current tile landed
    else      __builtin_amdgcn_s_wait_asynccnt(0);
#endif
    __syncthreads();

    FragAB a[2], b[4];
#pragma unroll
    for (int mi = 0; mi < 2; ++mi) {
      // A 16-bit fragment: lane(m=lm,kh): K = 8*kh+e (e<8) / 8*kh+e+8 (e>=8)
      const unsigned short* ar = &Atile[buf][(wr * 32 + mi * 16 + lm) * 32];
      a[mi].q[0] = *(const u32x4*)(ar + 8 * kh);
      a[mi].q[1] = *(const u32x4*)(ar + 16 + 8 * kh);
    }
#pragma unroll
    for (int ni = 0; ni < 4; ++ni) {
      // B fragment: lane(n=lm,kh): K = 8*kh+e (e<8) / 8*kh+e+8 (e>=8)
      const unsigned short* br = &Btile[buf][(wc * 64 + ni * 16 + lm) * 32];
      b[ni].q[0] = *(const u32x4*)(br + 8 * kh);
      b[ni].q[1] = *(const u32x4*)(br + 16 + 8 * kh);
    }
#pragma unroll
    for (int mi = 0; mi < 2; ++mi)
#pragma unroll
      for (int ni = 0; ni < 4; ++ni)
        acc[mi][ni] = __builtin_amdgcn_wmma_f32_16x16x32_bf16(
            false, a[mi].v, false, b[ni].v, (short)0, acc[mi][ni], false, false);

    __syncthreads();                    // safe to overwrite 'buf' next iter
    buf ^= 1;
  }

#pragma unroll
  for (int ni = 0; ni < 4; ++ni) {
    const int n = cb0 + wc * 64 + ni * 16 + lm;
    const float bv = bias ? bias[n] : 0.0f;
#pragma unroll
    for (int mi = 0; mi < 2; ++mi) {
#pragma unroll
      for (int v = 0; v < 8; ++v) {
        int r = rb0 + wr * 32 + mi * 16 + v + 8 * kh;  // C row = v + 8*(lane>=16)
        if (r < M) {
          float val = acc[mi][ni][v] + bv;
          if (RELU) val = fmaxf(val, 0.0f);
          if (WRITE_F32)  Cf[(size_t)r * N + n] = val;
          if (WRITE_BF16) Cb[(size_t)r * N + n] = f2bf(val);
        }
      }
    }
  }
}

// ---------------------------------------------------------------------------
// Flash self-attention. 1 wave = 16 query rows; keys streamed 32 at a time.
// Qh (B,NH,NQP,HD) bf16, Kh (B,NH,NQP,HD) bf16 zero-padded, Vt (B,NH,HD,NQP).
// ---------------------------------------------------------------------------
__global__ __launch_bounds__(128)
void flash_attn_wmma(const unsigned short* __restrict__ Qh,
                     const unsigned short* __restrict__ Kh,
                     const unsigned short* __restrict__ Vt,
                     unsigned short* __restrict__ Obf)   // (B,NQ,D) bf16
{
  __shared__ float pbuf[4][16 * 32];
  const int lane = threadIdx.x & 31;
  const int wave = threadIdx.x >> 5;
  const int lm = lane & 15, khf = lane >> 4;

  int blk = blockIdx.x;                 // B*NH*16 blocks
  const int qb = blk & 15; blk >>= 4;
  const int h = blk % NHH; const int b = blk / NHH;
  const int q0 = qb * 64 + wave * 16;

  const unsigned short* qbase = Qh + (size_t)(b * NHH + h) * NQP * HDD;
  const unsigned short* kbase = Kh + (size_t)(b * NHH + h) * NQP * HDD;
  const unsigned short* vbase = Vt + (size_t)(b * NHH + h) * HDD * NQP;

  FragAB aq;
  {
    int r = q0 + lm; if (r >= NQV) r = NQV - 1;
    const unsigned short* p = qbase + (size_t)r * HDD;
    aq.q[0] = *(const u32x4*)(p + 16 * khf);
    aq.q[1] = *(const u32x4*)(p + 16 * khf + 8);
  }

  f32x8 o0 = (f32x8){0.f,0.f,0.f,0.f,0.f,0.f,0.f,0.f};
  f32x8 o1 = o0;
  float mrow[8], lrow[8];
#pragma unroll
  for (int v = 0; v < 8; ++v) { mrow[v] = -__builtin_inff(); lrow[v] = 0.f; }

  const float sc = 0.17677669529663688f;  // 1/sqrt(32)
  float* P = pbuf[wave];
  const f32x8 zro = (f32x8){0.f,0.f,0.f,0.f,0.f,0.f,0.f,0.f};

  for (int kb = 0; kb < NQP; kb += 32) {
    FragAB bk0, bk1;
    {
      const unsigned short* p0 = kbase + (size_t)(kb + lm) * HDD + 16 * khf;
      bk0.q[0] = *(const u32x4*)(p0);  bk0.q[1] = *(const u32x4*)(p0 + 8);
      const unsigned short* p1 = kbase + (size_t)(kb + 16 + lm) * HDD + 16 * khf;
      bk1.q[0] = *(const u32x4*)(p1);  bk1.q[1] = *(const u32x4*)(p1 + 8);
    }
    f32x8 s0 = __builtin_amdgcn_wmma_f32_16x16x32_bf16(
        false, aq.v, false, bk0.v, (short)0, zro, false, false);
    f32x8 s1 = __builtin_amdgcn_wmma_f32_16x16x32_bf16(
        false, aq.v, false, bk1.v, (short)0, zro, false, false);

    const bool v0 = (kb + lm) < NQV;
    const bool v1 = (kb + 16 + lm) < NQV;
#pragma unroll
    for (int v = 0; v < 8; ++v) {
      float a0 = v0 ? s0[v] * sc : -__builtin_inff();
      float a1 = v1 ? s1[v] * sc : -__builtin_inff();
      float mx = fmaxf(a0, a1);
      mx = fmaxf(mx, __shfl_xor(mx, 1));
      mx = fmaxf(mx, __shfl_xor(mx, 2));
      mx = fmaxf(mx, __shfl_xor(mx, 4));
      mx = fmaxf(mx, __shfl_xor(mx, 8));
      float mnew = fmaxf(mrow[v], mx);
      float scal = __expf(mrow[v] - mnew);
      float p0 = __expf(a0 - mnew);
      float p1 = __expf(a1 - mnew);
      float ps = p0 + p1;
      ps += __shfl_xor(ps, 1); ps += __shfl_xor(ps, 2);
      ps += __shfl_xor(ps, 4); ps += __shfl_xor(ps, 8);
      lrow[v] = lrow[v] * scal + ps;
      mrow[v] = mnew;
      o0[v] *= scal; o1[v] *= scal;
      P[(v + 8 * khf) * 32 + lm]      = p0;   // stage P for A-layout repack
      P[(v + 8 * khf) * 32 + 16 + lm] = p1;
    }
    __builtin_amdgcn_wave_barrier();

    FragAB ap;
#pragma unroll
    for (int e = 0; e < 16; ++e) {
      int k = 16 * khf + e;
      ap.u[e] = f2bf(P[lm * 32 + k]);
    }
    __builtin_amdgcn_wave_barrier();

    FragAB bv0, bv1;
    {
      const unsigned short* p0 = vbase + (size_t)lm * NQP + kb + 16 * khf;
      bv0.q[0] = *(const u32x4*)(p0);  bv0.q[1] = *(const u32x4*)(p0 + 8);
      const unsigned short* p1 = vbase + (size_t)(16 + lm) * NQP + kb + 16 * khf;
      bv1.q[0] = *(const u32x4*)(p1);  bv1.q[1] = *(const u32x4*)(p1 + 8);
    }
    o0 = __builtin_amdgcn_wmma_f32_16x16x32_bf16(
        false, ap.v, false, bv0.v, (short)0, o0, false, false);
    o1 = __builtin_amdgcn_wmma_f32_16x16x32_bf16(
        false, ap.v, false, bv1.v, (short)0, o1, false, false);
  }

#pragma unroll
  for (int v = 0; v < 8; ++v) {
    int r = q0 + v + 8 * khf;
    if (r < NQV) {
      float inv = 1.0f / lrow[v];
      Obf[((size_t)b * NQV + r) * DM + h * HDD + lm]      = f2bf(o0[v] * inv);
      Obf[((size_t)b * NQV + r) * DM + h * HDD + 16 + lm] = f2bf(o1[v] * inv);
    }
  }
}

// ---------------------------------------------------------------------------
// Residual + LayerNorm. One wave per 256-wide row.
// ---------------------------------------------------------------------------
template<int WRITE_BF, int WRITE_POSBF>
__global__ __launch_bounds__(256)
void add_ln(const float* __restrict__ X, const float* __restrict__ R,
            const float* __restrict__ g, const float* __restrict__ bt,
            const float* __restrict__ pos,
            float* __restrict__ Yf, unsigned short* __restrict__ Yb,
            unsigned short* __restrict__ Ypb, int Mrows)
{
  int row = blockIdx.x * 8 + (threadIdx.x >> 5);
  if (row >= Mrows) return;
  int lane = threadIdx.x & 31;
  const float* xr = X + (size_t)row * DM;
  const float* rr = R + (size_t)row * DM;
  float vals[8]; float s = 0.f;
#pragma unroll
  for (int i = 0; i < 8; ++i) { vals[i] = xr[lane + 32 * i] + rr[lane + 32 * i]; s += vals[i]; }
#pragma unroll
  for (int off = 1; off < 32; off <<= 1) s += __shfl_xor(s, off);
  float mu = s * (1.0f / DM);
  float vs = 0.f;
#pragma unroll
  for (int i = 0; i < 8; ++i) { float d = vals[i] - mu; vs += d * d; }
#pragma unroll
  for (int off = 1; off < 32; off <<= 1) vs += __shfl_xor(vs, off);
  float inv = rsqrtf(vs * (1.0f / DM) + 1e-5f);
#pragma unroll
  for (int i = 0; i < 8; ++i) {
    int c = lane + 32 * i;
    float y = (vals[i] - mu) * inv * g[c] + bt[c];
    Yf[(size_t)row * DM + c] = y;
    if (WRITE_BF)    Yb[(size_t)row * DM + c]  = f2bf(y);
    if (WRITE_POSBF) Ypb[(size_t)row * DM + c] = f2bf(y + pos[(size_t)row * DM + c]);
  }
}

// --------------------------- small helper kernels --------------------------
__global__ void wt_transpose_bf(const float* __restrict__ W,
                                unsigned short* __restrict__ Wt, int K, int N)
{
  size_t i = (size_t)blockIdx.x * 256 + threadIdx.x;
  if (i >= (size_t)K * N) return;
  int n = (int)(i / K), k = (int)(i % K);
  Wt[i] = f2bf(W[(size_t)k * N + n]);
}

__global__ void cvt_qp(const float* __restrict__ q, const float* __restrict__ qp,
                       unsigned short* __restrict__ qp_bf,
                       unsigned short* __restrict__ q_bf, size_t n)
{
  size_t i = (size_t)blockIdx.x * 256 + threadIdx.x;
  if (i >= n) return;
  float a = q[i];
  qp_bf[i] = f2bf(a + qp[i]);
  q_bf[i]  = f2bf(a);
}

__global__ void cvt_bf(const float* __restrict__ a, unsigned short* __restrict__ o, size_t n)
{
  size_t i = (size_t)blockIdx.x * 256 + threadIdx.x;
  if (i < n) o[i] = f2bf(a[i]);
}

// (B,NQ,NH*HD) f32 -> (B,NH,NQP,HD) bf16, zero padded to NQP keys
__global__ void rearrange_heads(const float* __restrict__ src, unsigned short* __restrict__ dst)
{
  size_t i = (size_t)blockIdx.x * 256 + threadIdx.x;
  if (i >= (size_t)BQ * NHH * NQP * HDD) return;
  int hd = (int)(i % HDD); size_t t = i / HDD;
  int key = (int)(t % NQP); t /= NQP;
  int h = (int)(t % NHH); int b = (int)(t / NHH);
  float v = (key < NQV) ? src[((size_t)b * NQV + key) * DM + h * HDD + hd] : 0.f;
  dst[i] = f2bf(v);
}

// (B,NQ,NH*HD) f32 -> (B,NH,HD,NQP) bf16 (transposed for P@V B-fragments)
__global__ void rearrange_vt(const float* __restrict__ src, unsigned short* __restrict__ dst)
{
  size_t i = (size_t)blockIdx.x * 256 + threadIdx.x;
  if (i >= (size_t)BQ * NHH * HDD * NQP) return;
  int key = (int)(i % NQP); size_t t = i / NQP;
  int hd = (int)(t % HDD); t /= HDD;
  int h = (int)(t % NHH); int b = (int)(t / NHH);
  float v = (key < NQV) ? src[((size_t)b * NQV + key) * DM + h * HDD + hd] : 0.f;
  dst[i] = f2bf(v);
}

// softmax over 16 (L*P) per (b,q,h); logits (B*NQ, 128) -> aw (B*NQ*NH, 16)
__global__ void softmax_lp(const float* __restrict__ logits, float* __restrict__ aw)
{
  int i = blockIdx.x * 256 + threadIdx.x;
  if (i >= BQ * NQV * NHH) return;
  const float* p = logits + (size_t)(i / NHH) * 128 + (i % NHH) * 16;
  float mx = -__builtin_inff();
#pragma unroll
  for (int j = 0; j < 16; ++j) mx = fmaxf(mx, p[j]);
  float e[16]; float s = 0.f;
#pragma unroll
  for (int j = 0; j < 16; ++j) { e[j] = __expf(p[j] - mx); s += e[j]; }
  float inv = 1.0f / s;
  float* o = aw + (size_t)i * 16;
#pragma unroll
  for (int j = 0; j < 16; ++j) o[j] = e[j] * inv;
}

// deformable bilinear sampler; 1 wave per (b,q,h), lane = channel (HD=32)
__global__ __launch_bounds__(256)
void ms_deform_sample(const unsigned short* __restrict__ value, // (B,S,NH,HD) bf16
                      const float* __restrict__ off,            // (B*NQ,256)
                      const float* __restrict__ aw,             // (B*NQ*NH,16)
                      const float* __restrict__ refp,           // (B,NQ,L,4)
                      unsigned short* __restrict__ outb)        // (B,NQ,D) bf16
{
  int w = blockIdx.x * 8 + (threadIdx.x >> 5);
  if (w >= BQ * NQV * NHH) return;
  int lane = threadIdx.x & 31;
  int h = w % NHH; int bq = w / NHH; int b = bq / NQV;
  const int Hs[4] = {128, 64, 32, 16};
  const int St[4] = {0, 16384, 20480, 21504};
  float acc = 0.f;
  const float* offp = off + (size_t)bq * 256 + h * 32;
  const float* awp  = aw + (size_t)w * 16;
#pragma unroll
  for (int l = 0; l < 4; ++l) {
    const float* rp = refp + ((size_t)bq * 4 + l) * 4;
    float cx = rp[0], cy = rp[1], cw = rp[2], chh = rp[3];
    int Hl = Hs[l], Wl = Hs[l];
    const unsigned short* vb = value + ((size_t)b * STOT + St[l]) * DM + h * HDD + lane;
#pragma unroll
    for (int p = 0; p < 4; ++p) {
      float ox = offp[(l * 4 + p) * 2 + 0], oy = offp[(l * 4 + p) * 2 + 1];
      float lx = (cx + ox * 0.125f * cw) * Wl - 0.5f;
      float ly = (cy + oy * 0.125f * chh) * Hl - 0.5f;
      float fx = floorf(lx), fy = floorf(ly);
      float tx = lx - fx, ty = ly - fy;
      int x0 = (int)fx, y0 = (int)fy;
      float a = awp[l * 4 + p];
      float sum = 0.f;
#pragma unroll
      for (int dy = 0; dy < 2; ++dy)
#pragma unroll
        for (int dx = 0; dx < 2; ++dx) {
          int xi = x0 + dx, yi = y0 + dy;
          bool valid = (xi >= 0) & (xi < Wl) & (yi >= 0) & (yi < Hl);
          int xc = min(max(xi, 0), Wl - 1), yc = min(max(yi, 0), Hl - 1);
          float wgt = (dx ? tx : 1.f - tx) * (dy ? ty : 1.f - ty);
          float vv = valid ? bf2f(vb[(size_t)(yc * Wl + xc) * DM]) : 0.f;
          sum += wgt * vv;
        }
      acc += a * sum;
    }
  }
  outb[(size_t)bq * DM + h * HDD + lane] = f2bf(acc);
}

// ---------------------------------------------------------------------------
extern "C" void kernel_launch(void* const* d_in, const int* in_sizes, int n_in,
                              void* d_out, int out_size, void* d_ws, size_t ws_size,
                              hipStream_t stream) {
  const float* query  = (const float*)d_in[0];
  const float* qpos   = (const float*)d_in[1];
  const float* refp   = (const float*)d_in[2];
  const float* flat   = (const float*)d_in[3];
  const float* Wq = (const float*)d_in[6];  const float* bq = (const float*)d_in[7];
  const float* Wk = (const float*)d_in[8];  const float* bk = (const float*)d_in[9];
  const float* Wv = (const float*)d_in[10]; const float* bv = (const float*)d_in[11];
  const float* Wo = (const float*)d_in[12]; const float* bo = (const float*)d_in[13];
  const float* g1 = (const float*)d_in[14]; const float* bt1 = (const float*)d_in[15];
  const float* Woff = (const float*)d_in[16]; const float* boff = (const float*)d_in[17];
  const float* Watt = (const float*)d_in[18]; const float* batt = (const float*)d_in[19];
  const float* Wval = (const float*)d_in[20]; const float* bval = (const float*)d_in[21];
  const float* Wco  = (const float*)d_in[22]; const float* bco  = (const float*)d_in[23];
  const float* g2 = (const float*)d_in[24]; const float* bt2 = (const float*)d_in[25];
  const float* W1 = (const float*)d_in[26]; const float* b1 = (const float*)d_in[27];
  const float* W2 = (const float*)d_in[28]; const float* b2 = (const float*)d_in[29];
  const float* g3 = (const float*)d_in[30]; const float* bt3 = (const float*)d_in[31];
  float* out = (float*)d_out;

  char* p = (char*)d_ws;
  auto alloc = [&](size_t bytes) -> char* {
    char* r = p; p += (bytes + 255) & ~(size_t)255; return r;
  };
  typedef unsigned short us;
  us* wq_t  = (us*)alloc((size_t)256 * 256 * 2);
  us* wk_t  = (us*)alloc((size_t)256 * 256 * 2);
  us* wv_t  = (us*)alloc((size_t)256 * 256 * 2);
  us* wo_t  = (us*)alloc((size_t)256 * 256 * 2);
  us* wval_t= (us*)alloc((size_t)256 * 256 * 2);
  us* wco_t = (us*)alloc((size_t)256 * 256 * 2);
  us* woff_t= (us*)alloc((size_t)256 * 256 * 2);
  us* watt_t= (us*)alloc((size_t)128 * 256 * 2);
  us* w1_t  = (us*)alloc((size_t)1024 * 256 * 2);
  us* w2_t  = (us*)alloc((size_t)256 * 1024 * 2);
  us* qp_bf = (us*)alloc((size_t)M1 * DM * 2);
  us* q_bf  = (us*)alloc((size_t)M1 * DM * 2);
  us* flat_bf = (us*)alloc((size_t)MV * DM * 2);
  float* q_f = (float*)alloc((size_t)M1 * DM * 4);
  float* k_f = (float*)alloc((size_t)M1 * DM * 4);
  float* v_f = (float*)alloc((size_t)M1 * DM * 4);
  us* qh = (us*)alloc((size_t)BQ * NHH * NQP * HDD * 2);
  us* kh = (us*)alloc((size_t)BQ * NHH * NQP * HDD * 2);
  us* vt = (us*)alloc((size_t)BQ * NHH * HDD * NQP * 2);
  us* attn_bf = (us*)alloc((size_t)M1 * DM * 2);
  float* sa_f = (float*)alloc((size_t)M1 * DM * 4);
  float* x_f  = (float*)alloc((size_t)M1 * DM * 4);
  us* q2_bf   = (us*)alloc((size_t)M1 * DM * 2);
  us* value_bf= (us*)alloc((size_t)MV * DM * 2);
  float* off_f= (float*)alloc((size_t)M1 * DM * 4);
  float* lg_f = (float*)alloc((size_t)M1 * 128 * 4);
  float* aw_f = (float*)alloc((size_t)M1 * NHH * 16 * 4);
  us* samp_bf = (us*)alloc((size_t)M1 * DM * 2);
  float* ca_f = (float*)alloc((size_t)M1 * DM * 4);
  float* y_f  = (float*)alloc((size_t)M1 * DM * 4);
  us* y_bf    = (us*)alloc((size_t)M1 * DM * 2);
  us* h_bf    = (us*)alloc((size_t)M1 * DFFN * 2);
  float* ffn_f= (float*)alloc((size_t)M1 * DM * 4);

  auto wtgrid = [](int K, int N) { return dim3(((size_t)K * N + 255) / 256); };
  // weights -> (N,K) bf16
  wt_transpose_bf<<<wtgrid(256,256), 256, 0, stream>>>(Wq,  wq_t,  256, 256);
  wt_transpose_bf<<<wtgrid(256,256), 256, 0, stream>>>(Wk,  wk_t,  256, 256);
  wt_transpose_bf<<<wtgrid(256,256), 256, 0, stream>>>(Wv,  wv_t,  256, 256);
  wt_transpose_bf<<<wtgrid(256,256), 256, 0, stream>>>(Wo,  wo_t,  256, 256);
  wt_transpose_bf<<<wtgrid(256,256), 256, 0, stream>>>(Wval,wval_t,256, 256);
  wt_transpose_bf<<<wtgrid(256,256), 256, 0, stream>>>(Wco, wco_t, 256, 256);
  wt_transpose_bf<<<wtgrid(256,256), 256, 0, stream>>>(Woff,woff_t,256, 256);
  wt_transpose_bf<<<wtgrid(256,128), 256, 0, stream>>>(Watt,watt_t,256, 128);
  wt_transpose_bf<<<wtgrid(256,1024),256, 0, stream>>>(W1,  w1_t,  256, 1024);
  wt_transpose_bf<<<wtgrid(1024,256),256, 0, stream>>>(W2,  w2_t,  1024, 256);

  // activation converts
  size_t nq_elems = (size_t)M1 * DM;
  cvt_qp<<<(nq_elems + 255) / 256, 256, 0, stream>>>(query, qpos, qp_bf, q_bf, nq_elems);
  size_t nf = (size_t)MV * DM;
  cvt_bf<<<(nf + 255) / 256, 256, 0, stream>>>(flat, flat_bf, nf);

  dim3 blk(256);
  dim3 g_256(256 / 128, (M1 + 127) / 128);   // N=256 GEMMs on 8000 rows
  dim3 g_128(1,        (M1 + 127) / 128);    // N=128
  dim3 g_dff(DFFN / 128, (M1 + 127) / 128);  // N=1024
  dim3 g_val(256 / 128, MV / 128);           // value projection

  // value projection (biggest GEMM) -> bf16, L2-resident for the sampler
  gemm_bf16_wmma<0,0,1><<<g_val, blk, 0, stream>>>(flat_bf, wval_t, bval,
      nullptr, value_bf, MV, 256, 256);

  // Q/K/V projections
  gemm_bf16_wmma<0,1,0><<<g_256, blk, 0, stream>>>(qp_bf, wq_t, bq, q_f, nullptr, M1, 256, 256);
  gemm_bf16_wmma<0,1,0><<<g_256, blk, 0, stream>>>(qp_bf, wk_t, bk, k_f, nullptr, M1, 256, 256);
  gemm_bf16_wmma<0,1,0><<<g_256, blk, 0, stream>>>(q_bf,  wv_t, bv, v_f, nullptr, M1, 256, 256);

  size_t nheads = (size_t)BQ * NHH * NQP * HDD;
  rearrange_heads<<<(nheads + 255) / 256, 256, 0, stream>>>(q_f, qh);
  rearrange_heads<<<(nheads + 255) / 256, 256, 0, stream>>>(k_f, kh);
  rearrange_vt  <<<(nheads + 255) / 256, 256, 0, stream>>>(v_f, vt);

  flash_attn_wmma<<<BQ * NHH * 16, 128, 0, stream>>>(qh, kh, vt, attn_bf);

  gemm_bf16_wmma<0,1,0><<<g_256, blk, 0, stream>>>(attn_bf, wo_t, bo, sa_f, nullptr, M1, 256, 256);
  add_ln<0,1><<<(M1 + 7) / 8, 256, 0, stream>>>(query, sa_f, g1, bt1, qpos,
      x_f, nullptr, q2_bf, M1);

  // deformable cross attention
  gemm_bf16_wmma<0,1,0><<<g_256, blk, 0, stream>>>(q2_bf, woff_t, boff, off_f, nullptr, M1, 256, 256);
  gemm_bf16_wmma<0,1,0><<<g_128, blk, 0, stream>>>(q2_bf, watt_t, batt, lg_f,  nullptr, M1, 128, 256);
  softmax_lp<<<(BQ * NQV * NHH + 255) / 256, 256, 0, stream>>>(lg_f, aw_f);
  ms_deform_sample<<<(BQ * NQV * NHH + 7) / 8, 256, 0, stream>>>(value_bf, off_f, aw_f, refp, samp_bf);
  gemm_bf16_wmma<0,1,0><<<g_256, blk, 0, stream>>>(samp_bf, wco_t, bco, ca_f, nullptr, M1, 256, 256);
  add_ln<1,0><<<(M1 + 7) / 8, 256, 0, stream>>>(x_f, ca_f, g2, bt2, nullptr,
      y_f, y_bf, nullptr, M1);

  // FFN
  gemm_bf16_wmma<1,0,1><<<g_dff, blk, 0, stream>>>(y_bf, w1_t, b1, nullptr, h_bf, M1, DFFN, 256);
  gemm_bf16_wmma<0,1,0><<<g_256, blk, 0, stream>>>(h_bf, w2_t, b2, ffn_f, nullptr, M1, 256, DFFN);
  add_ln<0,0><<<(M1 + 7) / 8, 256, 0, stream>>>(y_f, ffn_f, g3, bt3, nullptr,
      out, nullptr, nullptr, M1);

  (void)in_sizes; (void)n_in; (void)out_size; (void)ws_size;
}
